// Attention_37864431681993
// MI455X (gfx1250) — matrix operand order repaired and checked by reading, not compiled, not run on hardware
//
#include <hip/hip_runtime.h>

#define L_DIM 2048
#define B_DIM 8
#define H_DIM 1024

typedef __attribute__((ext_vector_type(16))) __bf16 bf16x16;
typedef __attribute__((ext_vector_type(8)))  __bf16 bf16x8;
typedef __attribute__((ext_vector_type(8)))  float  f32x8;

// ---------------------------------------------------------------------------
// 1) Convert X (L,B,H) fp32 -> Xbf (B,L,H) bf16, 8 elements per thread.
//    Reads 32B/lane (2x b128), writes 16B/lane (1x b128). Coalesced in h.
// ---------------------------------------------------------------------------
__global__ void __launch_bounds__(256)
attn_convert_bf16(const float* __restrict__ X, __bf16* __restrict__ Xbf) {
  size_t idx = ((size_t)blockIdx.x * blockDim.x + threadIdx.x) * 8;  // over L*B*H
  size_t h = idx & (H_DIM - 1);
  size_t t = idx >> 10;           // / H_DIM
  size_t b = t & (B_DIM - 1);
  size_t l = t >> 3;              // / B_DIM
  f32x8 v = *(const f32x8*)(X + idx);
  bf16x8 o;
#pragma unroll
  for (int i = 0; i < 8; i++) o[i] = (__bf16)v[i];
  *(bf16x8*)(Xbf + (b * L_DIM + l) * H_DIM + h) = o;
}

// ---------------------------------------------------------------------------
// 2) scores[b] = scale * Xb[b] @ Xb[b]^T via V_WMMA_F32_16X16X32_BF16.
//    Block = 256 threads = 8 waves laid out 4(q) x 2(k); each wave owns a
//    64x64 tile (4x4 of 16x16), so a block covers 256(q) x 128(k).
// ---------------------------------------------------------------------------
__global__ void __launch_bounds__(256)
attn_scores_wmma(const __bf16* __restrict__ Xbf, float* __restrict__ attn) {
  const int b    = blockIdx.z;
  const int lane = threadIdx.x & 31;
  const int wid  = threadIdx.x >> 5;
  const int half = lane >> 4;     // 0 or 1 (lane group)
  const int l16  = lane & 15;
  const int q0 = blockIdx.x * 256 + (wid >> 1) * 64;
  const int k0 = blockIdx.y * 128 + (wid & 1) * 64;
  const __bf16* Xb = Xbf + (size_t)b * L_DIM * H_DIM;

  f32x8 zero = {};
  f32x8 acc[4][4];
#pragma unroll
  for (int i = 0; i < 4; i++)
#pragma unroll
    for (int j = 0; j < 4; j++) acc[i][j] = zero;

#pragma unroll 1
  for (int h = 0; h < H_DIM; h += 32) {
    bf16x16 afrag[4], bfrag[4];
    // A-matrix 16x32 layout: lanes0-15 K=0..7,16..23; lanes16-31 K=8..15,24..31
#pragma unroll
    for (int i = 0; i < 4; i++) {
      const __bf16* rp = Xb + (size_t)(q0 + i * 16 + l16) * H_DIM + h + half * 8;
      union { bf16x16 v; bf16x8 p[2]; } u;
      u.p[0] = *(const bf16x8*)(rp);
      u.p[1] = *(const bf16x8*)(rp + 16);
      afrag[i] = u.v;
    }
    // B-matrix 32x16 layout: N = lane&15, lanes0-15 K=0..7,16..23; lanes16-31 K=8..15,24..31
#pragma unroll
    for (int j = 0; j < 4; j++) {
      const __bf16* rp = Xb + (size_t)(k0 + j * 16 + l16) * H_DIM + h + half * 8;
      union { bf16x16 v; bf16x8 p[2]; } u;
      u.p[0] = *(const bf16x8*)(rp);
      u.p[1] = *(const bf16x8*)(rp + 16);
      bfrag[j] = u.v;
    }
#pragma unroll
    for (int i = 0; i < 4; i++)
#pragma unroll
      for (int j = 0; j < 4; j++)
        acc[i][j] = __builtin_amdgcn_wmma_f32_16x16x32_bf16(
            false, afrag[i], false, bfrag[j], (short)0, acc[i][j], false, false);
  }

  // C/D layout: VGPR v -> M = v + half*8, N = l16
  const float scale = 0.03125f;  // 1/sqrt(1024)
  float* outp = attn + (size_t)b * L_DIM * L_DIM;
#pragma unroll
  for (int i = 0; i < 4; i++)
#pragma unroll
    for (int v = 0; v < 8; v++) {
      size_t row = (size_t)(q0 + i * 16 + half * 8 + v);
      float* rowp = outp + row * L_DIM;
#pragma unroll
      for (int j = 0; j < 4; j++)
        rowp[k0 + j * 16 + l16] = acc[i][j][v] * scale;
    }
}

// ---------------------------------------------------------------------------
// 3) Row softmax in place. One 256-thread block per row of 2048.
// ---------------------------------------------------------------------------
__global__ void __launch_bounds__(256)
attn_softmax_rows(float* __restrict__ attn) {
  __shared__ float red[256];
  const int t = threadIdx.x;
  float* p = attn + (size_t)blockIdx.x * L_DIM;

  float v[8];
  float m = -3.4e38f;
#pragma unroll
  for (int i = 0; i < 8; i++) { v[i] = p[t + i * 256]; m = fmaxf(m, v[i]); }
  red[t] = m; __syncthreads();
  for (int s = 128; s > 0; s >>= 1) {
    if (t < s) red[t] = fmaxf(red[t], red[t + s]);
    __syncthreads();
  }
  m = red[0]; __syncthreads();

  float sum = 0.f;
#pragma unroll
  for (int i = 0; i < 8; i++) { v[i] = __expf(v[i] - m); sum += v[i]; }
  red[t] = sum; __syncthreads();
  for (int s = 128; s > 0; s >>= 1) {
    if (t < s) red[t] += red[t + s];
    __syncthreads();
  }
  const float inv = 1.0f / red[0];
#pragma unroll
  for (int i = 0; i < 8; i++) p[t + i * 256] = v[i] * inv;
}

// ---------------------------------------------------------------------------
// 4) colsum[b][k] = sum_q attn[b][q][k]  (coalesced across k, deterministic)
// ---------------------------------------------------------------------------
__global__ void __launch_bounds__(256)
attn_colsum(const float* __restrict__ attn, float* __restrict__ colsum) {
  const int b = blockIdx.y;
  const int k = blockIdx.x * 256 + threadIdx.x;
  const float* p = attn + (size_t)b * L_DIM * L_DIM + k;
  float s = 0.f;
#pragma unroll 4
  for (int q = 0; q < L_DIM; q++) s += p[(size_t)q * L_DIM];
  colsum[b * L_DIM + k] = s;
}

// ---------------------------------------------------------------------------
// 5) context[b][h] = (1/L) * sum_k colsum[b][k] * X[k][b][h]  (fp32 X)
// ---------------------------------------------------------------------------
__global__ void __launch_bounds__(256)
attn_context_mean(const float* __restrict__ X, const float* __restrict__ colsum,
                  float* __restrict__ outctx) {
  const int b = blockIdx.y;
  const int h = blockIdx.x * 256 + threadIdx.x;
  const float* cs = colsum + b * L_DIM;
  float s = 0.f;
#pragma unroll 4
  for (int k = 0; k < L_DIM; k++)
    s += cs[k] * X[((size_t)k * B_DIM + b) * H_DIM + h];
  outctx[b * H_DIM + h] = s * (1.0f / L_DIM);
}

// ---------------------------------------------------------------------------
extern "C" void kernel_launch(void* const* d_in, const int* in_sizes, int n_in,
                              void* d_out, int out_size, void* d_ws, size_t ws_size,
                              hipStream_t stream) {
  const float* X = (const float*)d_in[0];

  // d_out = [context (B*H floats)] ++ [attn (B*L*L floats)]
  float* out_ctx = (float*)d_out;
  float* attn    = out_ctx + (size_t)B_DIM * H_DIM;

  // workspace: bf16 copy of Xb (32 MB) then colsum (64 KB)
  __bf16* Xbf   = (__bf16*)d_ws;
  float* colsum = (float*)((char*)d_ws +
                           (size_t)B_DIM * L_DIM * H_DIM * sizeof(__bf16));

  const unsigned cvt_blocks = (unsigned)(((size_t)L_DIM * B_DIM * H_DIM) / (256 * 8));

  attn_convert_bf16<<<dim3(cvt_blocks), 256, 0, stream>>>(X, Xbf);
  attn_scores_wmma<<<dim3(L_DIM / 256, L_DIM / 128, B_DIM), 256, 0, stream>>>(Xbf, attn);
  attn_softmax_rows<<<dim3(B_DIM * L_DIM), 256, 0, stream>>>(attn);
  attn_colsum<<<dim3(L_DIM / 256, B_DIM), 256, 0, stream>>>(attn, colsum);
  attn_context_mean<<<dim3(H_DIM / 256, B_DIM), 256, 0, stream>>>(X, colsum, out_ctx);
}
